// RelationLearner_56573309223582
// MI455X (gfx1250) — compile-verified, matrix operand order checked by reading, not run on hardware
//
#include <hip/hip_runtime.h>

typedef __attribute__((ext_vector_type(16))) __bf16 v16bf;
typedef __attribute__((ext_vector_type(8)))  float  v8f;

union Frag {
    unsigned u[8];
    v16bf    v;
};

// ---- f32 -> bf16 conversion -------------------------------------------------
// Software RNE fallback (used on host pass / non-gfx1250)
__device__ __forceinline__ unsigned short f2bf_sw(float f) {
    unsigned u = __float_as_uint(f);
    unsigned r = u + 0x7FFFu + ((u >> 16) & 1u);
    return (unsigned short)(r >> 16);
}

__device__ __forceinline__ unsigned pack2_sw(unsigned short lo, unsigned short hi) {
    return (unsigned)lo | ((unsigned)hi << 16);
}

#if defined(__gfx1250__)
// V_FMA_MIXLO_BF16 / V_FMA_MIXHI_BF16 (CDNA5 VOP3P opcodes 62/63):
// D[15:0]/[31:16].bf16 = cvt_rtne(fma(s0, 1.0, 0)) -- 1 VALU op per element.
__device__ __forceinline__ unsigned cvt2_bf16(float lo, float hi) {
    unsigned d;
    asm("v_fma_mixlo_bf16 %0, %1, 1.0, 0" : "=v"(d) : "v"(lo));
    asm("v_fma_mixhi_bf16 %0, %1, 1.0, 0" : "+v"(d) : "v"(hi));
    return d;
}
__device__ __forceinline__ unsigned short cvt1_bf16(float f) {
    unsigned d;
    asm("v_fma_mixlo_bf16 %0, %1, 1.0, 0" : "=v"(d) : "v"(f));
    return (unsigned short)d;
}
#else
__device__ __forceinline__ unsigned cvt2_bf16(float lo, float hi) {
    return pack2_sw(f2bf_sw(lo), f2bf_sw(hi));
}
__device__ __forceinline__ unsigned short cvt1_bf16(float f) { return f2bf_sw(f); }
#endif

__device__ __forceinline__ v8f wmma_bf16(const Frag& a, const Frag& b, v8f c) {
    return __builtin_amdgcn_wmma_f32_16x16x32_bf16(
        /*neg_a=*/false, a.v, /*neg_b=*/false, b.v,
        /*c_mod=*/(short)0, c, /*reuse_a=*/false, /*reuse_b=*/false);
}

// ---- fragment loaders -------------------------------------------------------
// A-matrix 16x32 bf16 layout (wave32): lane L holds row m = L&15;
// kb = (L<16)?0:8; VGPR v in 0..3 -> K = kb+2v (pair), v in 4..7 -> K = kb+16+2(v-4).

// A from row-major f32 source (row pointer already at row*ld + k0)
__device__ __forceinline__ void loadAfrag_f32(const float* row, int kb, Frag& a) {
    float4 q0 = *(const float4*)(row + kb);
    float4 q1 = *(const float4*)(row + kb + 4);
    float4 q2 = *(const float4*)(row + kb + 16);
    float4 q3 = *(const float4*)(row + kb + 20);
    a.u[0] = cvt2_bf16(q0.x, q0.y);
    a.u[1] = cvt2_bf16(q0.z, q0.w);
    a.u[2] = cvt2_bf16(q1.x, q1.y);
    a.u[3] = cvt2_bf16(q1.z, q1.w);
    a.u[4] = cvt2_bf16(q2.x, q2.y);
    a.u[5] = cvt2_bf16(q2.z, q2.w);
    a.u[6] = cvt2_bf16(q3.x, q3.y);
    a.u[7] = cvt2_bf16(q3.z, q3.w);
}

// A from row-major bf16 (ushort) source; K-pairs are contiguous -> 2x b128 loads
__device__ __forceinline__ void loadAfrag_bf16(const unsigned short* row, int kb, Frag& a) {
    uint4 q0 = *(const uint4*)(row + kb);        // K = kb..kb+7
    uint4 q1 = *(const uint4*)(row + kb + 16);   // K = kb+16..kb+23
    a.u[0] = q0.x; a.u[1] = q0.y; a.u[2] = q0.z; a.u[3] = q0.w;
    a.u[4] = q1.x; a.u[5] = q1.y; a.u[6] = q1.z; a.u[7] = q1.w;
}

// B fragments pre-swizzled: per tile, lane L owns dwords [L*8 .. L*8+7].
// Works for both global and LDS pointers (2x b128 / ds_load_b128).
__device__ __forceinline__ void loadBfrag(const unsigned* packed, int tile, int lane, Frag& b) {
    const uint4* p = (const uint4*)(packed + tile * 256 + lane * 8);
    uint4 q0 = p[0];
    uint4 q1 = p[1];
    b.u[0] = q0.x; b.u[1] = q0.y; b.u[2] = q0.z; b.u[3] = q0.w;
    b.u[4] = q1.x; b.u[5] = q1.y; b.u[6] = q1.z; b.u[7] = q1.w;
}

// ---- weight pre-swizzle -----------------------------------------------------
__device__ void packW(const float* W, int K, int Ncols, int NtilePad,
                      unsigned* dst, int tid, int nthreads) {
    int Kt = K / 32;
    int total = Kt * NtilePad * 256;
    for (int i = tid; i < total; i += nthreads) {
        int t    = i >> 8;
        int r    = i & 255;
        int lane = r >> 3;
        int v    = r & 7;
        int kt = t / NtilePad;
        int nt = t - kt * NtilePad;
        int n  = nt * 16 + (lane & 15);
        int kb = (lane < 16) ? 0 : 8;
        int k  = kt * 32 + kb + ((v < 4) ? (2 * v) : (16 + 2 * (v - 4)));
        float lo = (n < Ncols) ? W[(size_t)k * Ncols + n]       : 0.0f;
        float hi = (n < Ncols) ? W[(size_t)(k + 1) * Ncols + n] : 0.0f;
        dst[t * 256 + lane * 8 + v] = cvt2_bf16(lo, hi);
    }
}

__global__ void prep_weights(const float* W1, const float* W2, const float* W3, const float* W4,
                             unsigned* W1p, unsigned* W2p, unsigned* W3p, unsigned* W4p) {
    int tid = threadIdx.x;
    int nt  = blockDim.x;
    if      (blockIdx.x == 0) packW(W1, 256, 64, 4, W1p, tid, nt);
    else if (blockIdx.x == 1) packW(W2,  64, 64, 4, W2p, tid, nt);
    else if (blockIdx.x == 2) packW(W3, 128, 64, 4, W3p, tid, nt);
    else                      packW(W4,  64, 10, 1, W4p, tid, nt);
}

// ---- node encoder: h = relu(x@W1+b1)@W2+b2, stored bf16 [N,64] --------------
__global__ __launch_bounds__(256) void encode_nodes(
    const float* __restrict__ x, const float* __restrict__ b1, const float* __restrict__ b2,
    const unsigned* __restrict__ W1p, const unsigned* __restrict__ W2p,
    unsigned short* __restrict__ hB, int N) {

    __shared__ unsigned short ldsH[8 * 16 * 64];   // per-wave 16x64 bf16 tile (16KB)

    int lane = threadIdx.x & 31;
    int wave = threadIdx.x >> 5;
    int m    = lane & 15;
    int n    = m;
    int kb   = (lane < 16) ? 0 : 8;

    int row0 = (blockIdx.x * 8 + wave) * 16;
    int r    = row0 + m;
    int rc   = (r < N) ? r : (N - 1);   // clamp: EXEC stays all-ones for WMMA

    v8f zero = {};
    v8f acc[4] = {zero, zero, zero, zero};

    // GEMM1: [16,256] x [256,64] -- batch 4 B-frag loads before the 4 WMMAs
    const float* xrow = x + (size_t)rc * 256;
    for (int kt = 0; kt < 8; ++kt) {
        Frag a;
        loadAfrag_f32(xrow + kt * 32, kb, a);
        Frag b[4];
#pragma unroll
        for (int t = 0; t < 4; ++t) loadBfrag(W1p, kt * 4 + t, lane, b[t]);
#pragma unroll
        for (int t = 0; t < 4; ++t) acc[t] = wmma_bf16(a, b[t], acc[t]);
    }

    // bias + relu -> LDS bf16 tile (C-frag -> A-frag relayout; wave-private)
    unsigned short* myT = ldsH + wave * (16 * 64);
#pragma unroll
    for (int t = 0; t < 4; ++t) {
        float bias = b1[t * 16 + n];
#pragma unroll
        for (int v = 0; v < 8; ++v) {
            int mr = (lane < 16) ? v : (v + 8);
            float val = acc[t][v] + bias;
            val = val > 0.0f ? val : 0.0f;
            myT[mr * 64 + t * 16 + n] = cvt1_bf16(val);
        }
    }

    // GEMM2: [16,64] x [64,64] (same-wave LDS ordering; no barrier needed)
    v8f acc2[4] = {zero, zero, zero, zero};
    for (int kt = 0; kt < 2; ++kt) {
        Frag a;
        loadAfrag_bf16(myT + m * 64 + kt * 32, kb, a);
        Frag b[4];
#pragma unroll
        for (int t = 0; t < 4; ++t) loadBfrag(W2p, kt * 4 + t, lane, b[t]);
#pragma unroll
        for (int t = 0; t < 4; ++t) acc2[t] = wmma_bf16(a, b[t], acc2[t]);
    }

#pragma unroll
    for (int t = 0; t < 4; ++t) {
        float bias = b2[t * 16 + n];
#pragma unroll
        for (int v = 0; v < 8; ++v) {
            int mr = (lane < 16) ? v : (v + 8);
            int rr = row0 + mr;
            if (rr < N) {
                float val = acc2[t][v] + bias;
                hB[(size_t)rr * 64 + t * 16 + n] = cvt1_bf16(val);
            }
        }
    }
}

// ---- edge relation MLP + softmax/argmax (persistent, LDS-resident weights) --
__global__ __launch_bounds__(256) void edge_relations(
    const int* __restrict__ eidx, const unsigned short* __restrict__ hB,
    const float* __restrict__ b3, const float* __restrict__ b4,
    const unsigned* __restrict__ W3p, const unsigned* __restrict__ W4p,
    float* __restrict__ outType, float* __restrict__ outProbs, int E, int niter) {

    __shared__ unsigned       ldsW[18 * 256];      // W3: 16 tiles + W4: 2 tiles (18KB)
    __shared__ unsigned short ldsG[8 * 16 * 64];   // relu'd hidden per wave (16KB)
    __shared__ float          ldsL[8 * 16 * 16];   // logits per wave (8KB)

    // stage packed weights into LDS once per block
    {
        const uint4* s3 = (const uint4*)W3p;   // 1024 uint4
        uint4*       d3 = (uint4*)ldsW;
        for (int i = threadIdx.x; i < 1024; i += 256) d3[i] = s3[i];
        const uint4* s4 = (const uint4*)W4p;   // 128 uint4
        uint4*       d4 = (uint4*)(ldsW + 16 * 256);
        for (int i = threadIdx.x; i < 128; i += 256) d4[i] = s4[i];
    }
    __syncthreads();
    const unsigned* W3l = ldsW;
    const unsigned* W4l = ldsW + 16 * 256;

    int lane = threadIdx.x & 31;
    int wave = threadIdx.x >> 5;
    int m    = lane & 15;
    int n    = m;
    int kb   = (lane < 16) ? 0 : 8;

    unsigned short* myG = ldsG + wave * (16 * 64);
    float*          myL = ldsL + wave * 256;
    float bias4 = (n < 10) ? b4[n] : 0.0f;

    int tileStride = gridDim.x * 8;
    int tile       = blockIdx.x * 8 + wave;

    // preload first tile's edge indices (clamped so loads are always valid)
    int e  = tile * 16 + m;
    int ec = (e < E) ? e : (E - 1);
    int nodeR = eidx[ec];
    int nodeC = eidx[E + ec];

    v8f zero = {};

    for (int it = 0; it < niter; ++it) {
        int curTile = tile;
        int curR = nodeR;
        int curC = nodeC;

        // kick off next tile's index loads + prefetch its gather rows
        tile += tileStride;
        if (it + 1 < niter) {
            int en  = tile * 16 + m;
            int ecn = (en < E) ? en : (E - 1);
            nodeR = eidx[ecn];
            nodeC = eidx[E + ecn];
            __builtin_prefetch(hB + (size_t)nodeR * 64, 0, 0);
            __builtin_prefetch(hB + (size_t)nodeC * 64, 0, 0);
        }

        v8f acc[4] = {zero, zero, zero, zero};

        // GEMM: edge_features[16,128] x W3[128,64]; concat = node choice per K-tile
        for (int kt = 0; kt < 4; ++kt) {
            int node = (kt < 2) ? curR : curC;
            int kk   = (kt & 1) * 32;
            Frag a;
            loadAfrag_bf16(hB + (size_t)node * 64 + kk, kb, a);
            Frag b[4];
#pragma unroll
            for (int t = 0; t < 4; ++t) loadBfrag(W3l, kt * 4 + t, lane, b[t]);
#pragma unroll
            for (int t = 0; t < 4; ++t) acc[t] = wmma_bf16(a, b[t], acc[t]);
        }

        // bias + relu -> wave-private LDS tile
#pragma unroll
        for (int t = 0; t < 4; ++t) {
            float bias = b3[t * 16 + n];
#pragma unroll
            for (int v = 0; v < 8; ++v) {
                int mr = (lane < 16) ? v : (v + 8);
                float val = acc[t][v] + bias;
                val = val > 0.0f ? val : 0.0f;
                myG[mr * 64 + t * 16 + n] = cvt1_bf16(val);
            }
        }

        // GEMM: [16,64] x W4pad[64,16] (same-wave LDS ordering)
        v8f accL = zero;
        for (int kt = 0; kt < 2; ++kt) {
            Frag a;
            loadAfrag_bf16(myG + m * 64 + kt * 32, kb, a);
            Frag b;
            loadBfrag(W4l, kt, lane, b);
            accL = wmma_bf16(a, b, accL);
        }

        // logits -> wave-private LDS (row-major 16x16)
#pragma unroll
        for (int v = 0; v < 8; ++v) {
            int mr = (lane < 16) ? v : (v + 8);
            myL[mr * 16 + n] = accL[v] + bias4;
        }

        // softmax + argmax: lanes 0..15 each own one edge (row)
        if (lane < 16) {
            int ee = curTile * 16 + lane;
            if (ee < E) {
                const float* rowL = myL + lane * 16;
                float l[10];
#pragma unroll
                for (int j = 0; j < 10; ++j) l[j] = rowL[j];
                float mx = l[0];
                int   am = 0;
#pragma unroll
                for (int j = 1; j < 10; ++j) {
                    if (l[j] > mx) { mx = l[j]; am = j; }   // first-max == jnp.argmax
                }
                float ex[10];
                float s = 0.0f;
#pragma unroll
                for (int j = 0; j < 10; ++j) { ex[j] = expf(l[j] - mx); s += ex[j]; }
                float inv = 1.0f / s;
#pragma unroll
                for (int j = 0; j < 10; ++j) outProbs[(size_t)ee * 10 + j] = ex[j] * inv;
                outType[ee] = (float)am;
            }
        }
    }
}

// ---- launcher ---------------------------------------------------------------
extern "C" void kernel_launch(void* const* d_in, const int* in_sizes, int n_in,
                              void* d_out, int out_size, void* d_ws, size_t ws_size,
                              hipStream_t stream) {
    const float* x    = (const float*)d_in[0];
    const int*   eidx = (const int*)  d_in[1];
    const float* W1   = (const float*)d_in[2];
    const float* b1   = (const float*)d_in[3];
    const float* W2   = (const float*)d_in[4];
    const float* b2   = (const float*)d_in[5];
    const float* W3   = (const float*)d_in[6];
    const float* b3   = (const float*)d_in[7];
    const float* W4   = (const float*)d_in[8];
    const float* b4   = (const float*)d_in[9];

    int N = in_sizes[0] / 256;   // NODE_DIM = 256
    int E = in_sizes[1] / 2;     // edge_index is [2, E]

    // workspace layout: packed weights (64KB) then bf16 h [N,64]
    unsigned char* ws = (unsigned char*)d_ws;
    unsigned*       W1p = (unsigned*)(ws);            // 32 tiles * 1KB
    unsigned*       W2p = (unsigned*)(ws + 32768);    //  8 tiles
    unsigned*       W3p = (unsigned*)(ws + 40960);    // 16 tiles
    unsigned*       W4p = (unsigned*)(ws + 57344);    //  2 tiles
    unsigned short* hB  = (unsigned short*)(ws + 65536);

    float* outType  = (float*)d_out;          // E argmax values (as float)
    float* outProbs = (float*)d_out + E;      // E x 10 probabilities

    prep_weights<<<4, 256, 0, stream>>>(W1, W2, W3, W4, W1p, W2p, W3p, W4p);

    int nb1 = (N + 127) / 128;   // 8 waves * 16 rows per block
    encode_nodes<<<nb1, 256, 0, stream>>>(x, b1, b2, W1p, W2p, hB, N);

    // persistent edge kernel: uniform trip count so the one staging barrier is safe
    int tilesTotal = (E + 15) / 16;
    int nb2 = (tilesTotal + 7) / 8;
    if (nb2 > 1536) nb2 = 1536;
    int slots = nb2 * 8;
    int niter = (tilesTotal + slots - 1) / slots;
    edge_relations<<<nb2, 256, 0, stream>>>(eidx, hB, b3, b4, W3p, W4p,
                                            outType, outProbs, E, niter);
}
